// AttentionBlock_self_crosss_78752520340152
// MI455X (gfx1250) — matrix operand order repaired and checked by reading, not compiled, not run on hardware
//
#include <hip/hip_runtime.h>
#include <hip/hip_bf16.h>
#include <math.h>

// ---------------------------------------------------------------------------
// CDNA5 (gfx1250) wave32 WMMA attention block.
// All matrix math goes through v_wmma_f32_16x16x32_bf16 (fp32 accumulate).
// GEMMs: 16x64 output tiles, register double-buffered K-loop so WMMAs of
// step i overlap the fragment loads of step i+1 (partial s_wait_loadcnt).
// Flash attention: 32 queries/wave, K/V fragments shared by both query tiles.
// ---------------------------------------------------------------------------

typedef __bf16 bf16_t;
typedef __attribute__((ext_vector_type(16))) __bf16 v16bf;
typedef __attribute__((ext_vector_type(8)))  __bf16 v8bf;
typedef __attribute__((ext_vector_type(8)))  float  v8f;

#define WMMA_BF16(A, Bv, Cv) \
  __builtin_amdgcn_wmma_f32_16x16x32_bf16(false, (A), false, (Bv), (short)0, (Cv), false, false)

__device__ __forceinline__ bf16_t f2bf(float f) {
  unsigned u = __builtin_bit_cast(unsigned, f);
  u += 0x7FFFu + ((u >> 16) & 1u);          // round-to-nearest-even
  unsigned short h = (unsigned short)(u >> 16);
  return __builtin_bit_cast(bf16_t, h);
}

// A-fragment (16x32 MxK bf16), contiguous k order (same as B):
//   lane L (m = L&15): halves e=0..15 -> k = c0 + 16*laneHi + e
__device__ __forceinline__ v16bf load_frag_a(const bf16_t* __restrict__ base,
                                             int stride, int row, int col0, int laneHi) {
  const bf16_t* p = base + (size_t)row * stride + col0 + laneHi * 16;
  v8bf lo = *(const v8bf*)(p);
  v8bf hi = *(const v8bf*)(p + 8);
  return __builtin_shufflevector(lo, hi, 0,1,2,3,4,5,6,7,8,9,10,11,12,13,14,15);
}

// B-fragment (32x16 KxN bf16): lane L holds column n = L&15,
//   halves e=0..15 -> k = c0 + 16*laneHi + e.
__device__ __forceinline__ v16bf load_frag_b(const bf16_t* __restrict__ base,
                                             int stride, int row, int col0, int laneHi) {
  const bf16_t* p = base + (size_t)row * stride + col0 + laneHi * 16;
  v8bf lo = *(const v8bf*)(p);
  v8bf hi = *(const v8bf*)(p + 8);
  return __builtin_shufflevector(lo, hi, 0,1,2,3,4,5,6,7,8,9,10,11,12,13,14,15);
}

// ---------------------------------------------------------------------------
// Elementwise / normalization kernels
// ---------------------------------------------------------------------------

__global__ void f32_to_bf16_kernel(const float* __restrict__ src,
                                   bf16_t* __restrict__ dst, int n) {
  int i = blockIdx.x * blockDim.x + threadIdx.x;
  if (i < n) dst[i] = f2bf(src[i]);
}

// GroupNorm over [B,C,T] and write transposed bf16 [B,T,C].
__global__ void gn_transpose_kernel(const float* __restrict__ x,
                                    const float* __restrict__ gamma,
                                    const float* __restrict__ beta,
                                    bf16_t* __restrict__ outT,
                                    int C, int T, int groups) {
  const int b = blockIdx.x / groups;
  const int g = blockIdx.x % groups;
  const int cpg = C / groups;
  const int n = cpg * T;
  const float* xg = x + ((size_t)b * C + (size_t)g * cpg) * T;

  float s = 0.f, ss = 0.f;
  for (int i = threadIdx.x; i < n; i += blockDim.x) {
    float v = xg[i];
    s += v; ss += v * v;
  }
  __shared__ float rs[256];
  __shared__ float rq[256];
  rs[threadIdx.x] = s; rq[threadIdx.x] = ss;
  __syncthreads();
  for (int off = 128; off > 0; off >>= 1) {
    if ((int)threadIdx.x < off) {
      rs[threadIdx.x] += rs[threadIdx.x + off];
      rq[threadIdx.x] += rq[threadIdx.x + off];
    }
    __syncthreads();
  }
  float mu  = rs[0] / (float)n;
  float var = rq[0] / (float)n - mu * mu;
  float inv = rsqrtf(var + 1e-5f);

  for (int i = threadIdx.x; i < n; i += blockDim.x) {
    int cl = i / T, t = i - cl * T;
    int c = g * cpg + cl;
    float v = (xg[i] - mu) * inv * gamma[c] + beta[c];
    outT[((size_t)b * T + t) * C + c] = f2bf(v);
  }
}

// cond2_p[b,c,t] = w[c] * cond2[b,t] + bias[c]   (outer product)
__global__ void cond2_proj_kernel(const float* __restrict__ cond2,
                                  const float* __restrict__ w,
                                  const float* __restrict__ bias,
                                  float* __restrict__ out,
                                  int C, int T, int B) {
  size_t i = (size_t)blockIdx.x * blockDim.x + threadIdx.x;
  size_t total = (size_t)B * C * T;
  if (i >= total) return;
  int t = (int)(i % T);
  int c = (int)((i / T) % C);
  int b = (int)(i / ((size_t)T * C));
  out[i] = w[c] * cond2[(size_t)b * T + t] + bias[c];
}

// ---------------------------------------------------------------------------
// GEMM: D[o,t] = sum_c W[o,c]*XT[b,t,c] + bias[o], 16x64 tile per wave,
// scattered into q/k/v layouts.
// mode 0: qkv interleave (192 rows/head: q,k,v)     mode 1: kv halves (k then v)
// ---------------------------------------------------------------------------
__global__ void gemm_split_kernel(const bf16_t* __restrict__ W,    // [M,K]
                                  const bf16_t* __restrict__ XT,   // [B,N,K]
                                  const float*  __restrict__ bias, // [M]
                                  bf16_t* __restrict__ qT,         // [BH,N,ch] (mode0)
                                  bf16_t* __restrict__ kT,         // [BH,N,ch]
                                  bf16_t* __restrict__ Vv,         // [BH,ch,N]
                                  int M, int K, int N, int ch, float scale, int mode) {
  const int lane   = threadIdx.x & 31;
  const int nn     = lane & 15;
  const int laneHi = lane >> 4;
  const int o0 = blockIdx.y * 16;
  const int t0 = blockIdx.x * 64;
  const int b  = blockIdx.z;
  const bf16_t* xb = XT + (size_t)b * N * K;

  const v8f z = {0.f,0.f,0.f,0.f,0.f,0.f,0.f,0.f};
  v8f acc[4] = {z, z, z, z};

  // Register double buffering: loads for step i+1 are issued before the
  // WMMAs of step i, so waits become partial instead of wait-all.
  v16bf a  = load_frag_a(W, K, o0 + nn, 0, laneHi);
  v16bf b0 = load_frag_b(xb, K, t0 + nn,      0, laneHi);
  v16bf b1 = load_frag_b(xb, K, t0 + 16 + nn, 0, laneHi);
  v16bf b2 = load_frag_b(xb, K, t0 + 32 + nn, 0, laneHi);
  v16bf b3 = load_frag_b(xb, K, t0 + 48 + nn, 0, laneHi);
  for (int c0 = 32; c0 < K; c0 += 32) {
    // prefetch 2 K-steps ahead into near cache (global_prefetch_b8, WGP hint)
    __builtin_prefetch(W + (size_t)(o0 + nn) * K + c0 + 32, 0, 3);
    v16bf an  = load_frag_a(W, K, o0 + nn, c0, laneHi);
    v16bf bn0 = load_frag_b(xb, K, t0 + nn,      c0, laneHi);
    v16bf bn1 = load_frag_b(xb, K, t0 + 16 + nn, c0, laneHi);
    v16bf bn2 = load_frag_b(xb, K, t0 + 32 + nn, c0, laneHi);
    v16bf bn3 = load_frag_b(xb, K, t0 + 48 + nn, c0, laneHi);
    acc[0] = WMMA_BF16(a, b0, acc[0]);
    acc[1] = WMMA_BF16(a, b1, acc[1]);
    acc[2] = WMMA_BF16(a, b2, acc[2]);
    acc[3] = WMMA_BF16(a, b3, acc[3]);
    a = an; b0 = bn0; b1 = bn1; b2 = bn2; b3 = bn3;
  }
  acc[0] = WMMA_BF16(a, b0, acc[0]);
  acc[1] = WMMA_BF16(a, b1, acc[1]);
  acc[2] = WMMA_BF16(a, b2, acc[2]);
  acc[3] = WMMA_BF16(a, b3, acc[3]);

  const int H = 8;
#pragma unroll
  for (int i = 0; i < 4; i++) {
#pragma unroll
    for (int r = 0; r < 8; r++) {
      int m = r + 8 * laneHi;          // C-fragment row (ISA 7.12.2)
      int o = o0 + m;
      int t = t0 + i * 16 + nn;
      float val = acc[i][r] + bias[o];
      int h, j, kind;                  // 0=q, 1=k, 2=v
      if (mode == 0) {
        int rph = 3 * ch;
        h = o / rph; j = o % rph;
        kind = j / ch; j = j % ch;
      } else {
        int half = M >> 1;
        if (o < half) { kind = 1; h = o / ch;  j = o % ch; }
        else          { kind = 2; int oo = o - half; h = oo / ch; j = oo % ch; }
      }
      size_t bh = (size_t)b * H + h;
      if (kind == 0)      qT[(bh * N + t) * ch + j] = f2bf(val * scale);
      else if (kind == 1) kT[(bh * N + t) * ch + j] = f2bf(val * scale);
      else                Vv[(bh * ch + j) * N + t] = f2bf(val);
    }
  }
}

// ---------------------------------------------------------------------------
// GEMM with residual: out[b,o,t] = resid + W*XT + bias, optional scaled-Q out.
// 16x64 tile per wave, double-buffered K-loop.
// ---------------------------------------------------------------------------
__global__ void gemm_resid_kernel(const bf16_t* __restrict__ W,     // [M,K]
                                  const bf16_t* __restrict__ XT,    // [B,N,K]
                                  const float*  __restrict__ bias,  // [M]
                                  const float*  __restrict__ resid, // [B,M,N]
                                  float* __restrict__ out,          // [B,M,N]
                                  bf16_t* __restrict__ qT2,         // [BH,N,ch] or null
                                  int M, int K, int N, int ch, float scale) {
  const int lane   = threadIdx.x & 31;
  const int nn     = lane & 15;
  const int laneHi = lane >> 4;
  const int o0 = blockIdx.y * 16;
  const int t0 = blockIdx.x * 64;
  const int b  = blockIdx.z;
  const bf16_t* xb = XT + (size_t)b * N * K;

  const v8f z = {0.f,0.f,0.f,0.f,0.f,0.f,0.f,0.f};
  v8f acc[4] = {z, z, z, z};

  v16bf a  = load_frag_a(W, K, o0 + nn, 0, laneHi);
  v16bf b0 = load_frag_b(xb, K, t0 + nn,      0, laneHi);
  v16bf b1 = load_frag_b(xb, K, t0 + 16 + nn, 0, laneHi);
  v16bf b2 = load_frag_b(xb, K, t0 + 32 + nn, 0, laneHi);
  v16bf b3 = load_frag_b(xb, K, t0 + 48 + nn, 0, laneHi);
  for (int c0 = 32; c0 < K; c0 += 32) {
    __builtin_prefetch(W + (size_t)(o0 + nn) * K + c0 + 32, 0, 3);
    v16bf an  = load_frag_a(W, K, o0 + nn, c0, laneHi);
    v16bf bn0 = load_frag_b(xb, K, t0 + nn,      c0, laneHi);
    v16bf bn1 = load_frag_b(xb, K, t0 + 16 + nn, c0, laneHi);
    v16bf bn2 = load_frag_b(xb, K, t0 + 32 + nn, c0, laneHi);
    v16bf bn3 = load_frag_b(xb, K, t0 + 48 + nn, c0, laneHi);
    acc[0] = WMMA_BF16(a, b0, acc[0]);
    acc[1] = WMMA_BF16(a, b1, acc[1]);
    acc[2] = WMMA_BF16(a, b2, acc[2]);
    acc[3] = WMMA_BF16(a, b3, acc[3]);
    a = an; b0 = bn0; b1 = bn1; b2 = bn2; b3 = bn3;
  }
  acc[0] = WMMA_BF16(a, b0, acc[0]);
  acc[1] = WMMA_BF16(a, b1, acc[1]);
  acc[2] = WMMA_BF16(a, b2, acc[2]);
  acc[3] = WMMA_BF16(a, b3, acc[3]);

#pragma unroll
  for (int i = 0; i < 4; i++) {
#pragma unroll
    for (int r = 0; r < 8; r++) {
      int m = r + 8 * laneHi;
      int o = o0 + m;
      int t = t0 + i * 16 + nn;
      size_t idx = ((size_t)b * M + o) * N + t;
      float val = acc[i][r] + bias[o] + resid[idx];
      out[idx] = val;
      if (qT2) {
        int h = o / ch, c = o % ch;
        qT2[(((size_t)b * 8 + h) * N + t) * ch + c] = f2bf(val * scale);
      }
    }
  }
}

// ---------------------------------------------------------------------------
// Flash attention helper: online-softmax update of one 32x16 S^T tile pair.
// Rescales O accumulators, accumulates row sums, stages P^T (bf16) to LDS.
// ---------------------------------------------------------------------------
__device__ __forceinline__ void softmax_update(const v8f& sf0, const v8f& sf1,
                                               v8f& o0, v8f& o1, v8f& o2, v8f& o3,
                                               float& mrun, float& lrun,
                                               bf16_t* pst, int nn, int laneHi) {
  float tmax = -3.0e38f;
#pragma unroll
  for (int r = 0; r < 8; r++) tmax = fmaxf(tmax, fmaxf(sf0[r], sf1[r]));
  tmax = fmaxf(tmax, __shfl_xor(tmax, 16, 32));
  float mnew  = fmaxf(mrun, tmax);
  float alpha = __expf(mrun - mnew);
  lrun *= alpha;
#pragma unroll
  for (int r = 0; r < 8; r++) {
    o0[r] *= alpha; o1[r] *= alpha; o2[r] *= alpha; o3[r] *= alpha;
  }
  v8bf pv0, pv1;
  float psum = 0.f;
#pragma unroll
  for (int r = 0; r < 8; r++) {
    float p0 = __expf(sf0[r] - mnew);
    float p1 = __expf(sf1[r] - mnew);
    psum += p0 + p1;
    pv0[r] = f2bf(p0);
    pv1[r] = f2bf(p1);
  }
  psum += __shfl_xor(psum, 16, 32);
  lrun += psum;
  mrun  = mnew;
  // P^T stage: [t][s_local], s_local = f*16 + 8*laneHi + r  (aligned b128 ds)
  *(v8bf*)(pst + nn * 32 + laneHi * 8)      = pv0;
  *(v8bf*)(pst + nn * 32 + 16 + laneHi * 8) = pv1;
}

// ---------------------------------------------------------------------------
// Flash attention, one wave per (bh, 32-query tile pair).
//   qT [BH,Tq,64] (scaled), kT [BH,Tk,64] (scaled), V [BH,64,Tk]
//   outT [B,Tq,H*64] bf16
// S^T tiles (M=s, N=t) keep the softmax axis in the register/M dimension:
// reduction = 8 in-lane values + one shfl_xor(16). K/V fragments are shared
// by both query tiles: 16 WMMAs per 16 b128 K/V loads. V loads are issued
// before the softmax VALU work so they complete under it.
// ---------------------------------------------------------------------------
__global__ void flash_attn_kernel(const bf16_t* __restrict__ qT,
                                  const bf16_t* __restrict__ kT,
                                  const bf16_t* __restrict__ V,
                                  bf16_t* __restrict__ outT,
                                  int Tq, int Tk, int H) {
  const int CH = 64;
  const int lane   = threadIdx.x & 31;
  const int nn     = lane & 15;
  const int laneHi = lane >> 4;
  const int bh = blockIdx.y;
  const int t0 = blockIdx.x * 32;     // two 16-query tiles: t0, t0+16

  const bf16_t* qb = qT + (size_t)bh * Tq * CH;
  const bf16_t* kb = kT + (size_t)bh * Tk * CH;
  const bf16_t* vb = V  + (size_t)bh * CH * Tk;

  // Q B-fragments loaded once (K-dim = ch = 64 -> two fragments per tile).
  v16bf BqA0 = load_frag_b(qb, CH, t0 + nn,      0,  laneHi);
  v16bf BqA1 = load_frag_b(qb, CH, t0 + nn,      32, laneHi);
  v16bf BqB0 = load_frag_b(qb, CH, t0 + 16 + nn, 0,  laneHi);
  v16bf BqB1 = load_frag_b(qb, CH, t0 + 16 + nn, 32, laneHi);

  __shared__ __align__(16) bf16_t pstA[16 * 32];
  __shared__ __align__(16) bf16_t pstB[16 * 32];

  const v8f z = {0.f,0.f,0.f,0.f,0.f,0.f,0.f,0.f};
  v8f oaA0 = z, oaA1 = z, oaA2 = z, oaA3 = z;   // O tiles, c = 0,16,32,48
  v8f oaB0 = z, oaB1 = z, oaB2 = z, oaB3 = z;
  float mA = -3.0e38f, lA = 0.f;
  float mB = -3.0e38f, lB = 0.f;

  for (int s0 = 0; s0 < Tk; s0 += 32) {
    // K^T fragments for this 32-key step (shared by both query tiles).
    v16bf a00 = load_frag_a(kb, CH, s0 + nn,      0,  laneHi);
    v16bf a01 = load_frag_a(kb, CH, s0 + nn,      32, laneHi);
    v16bf a10 = load_frag_a(kb, CH, s0 + 16 + nn, 0,  laneHi);
    v16bf a11 = load_frag_a(kb, CH, s0 + 16 + nn, 32, laneHi);

    v8f sA0 = WMMA_BF16(a00, BqA0, z);
    sA0     = WMMA_BF16(a01, BqA1, sA0);
    v8f sA1 = WMMA_BF16(a10, BqA0, z);
    sA1     = WMMA_BF16(a11, BqA1, sA1);
    v8f sB0 = WMMA_BF16(a00, BqB0, z);
    sB0     = WMMA_BF16(a01, BqB1, sB0);
    v8f sB1 = WMMA_BF16(a10, BqB0, z);
    sB1     = WMMA_BF16(a11, BqB1, sB1);

    // V fragments for this key step (no dependence on softmax): issue early
    // so they complete underneath the exp/rescale VALU and the LDS staging.
    v16bf av0 = load_frag_a(vb, Tk, 0  + nn, s0, laneHi);
    v16bf av1 = load_frag_a(vb, Tk, 16 + nn, s0, laneHi);
    v16bf av2 = load_frag_a(vb, Tk, 32 + nn, s0, laneHi);
    v16bf av3 = load_frag_a(vb, Tk, 48 + nn, s0, laneHi);

    softmax_update(sA0, sA1, oaA0, oaA1, oaA2, oaA3, mA, lA, pstA, nn, laneHi);
    softmax_update(sB0, sB1, oaB0, oaB1, oaB2, oaB3, mB, lB, pstB, nn, laneHi);
    __syncthreads();

    v8bf plo = *(const v8bf*)(pstA + nn * 32 + laneHi * 16);
    v8bf phi = *(const v8bf*)(pstA + nn * 32 + laneHi * 16 + 8);
    v16bf BpA = __builtin_shufflevector(plo, phi,
                                        0,1,2,3,4,5,6,7,8,9,10,11,12,13,14,15);
    plo = *(const v8bf*)(pstB + nn * 32 + laneHi * 16);
    phi = *(const v8bf*)(pstB + nn * 32 + laneHi * 16 + 8);
    v16bf BpB = __builtin_shufflevector(plo, phi,
                                        0,1,2,3,4,5,6,7,8,9,10,11,12,13,14,15);

    oaA0 = WMMA_BF16(av0, BpA, oaA0);
    oaA1 = WMMA_BF16(av1, BpA, oaA1);
    oaA2 = WMMA_BF16(av2, BpA, oaA2);
    oaA3 = WMMA_BF16(av3, BpA, oaA3);
    oaB0 = WMMA_BF16(av0, BpB, oaB0);
    oaB1 = WMMA_BF16(av1, BpB, oaB1);
    oaB2 = WMMA_BF16(av2, BpB, oaB2);
    oaB3 = WMMA_BF16(av3, BpB, oaB3);
    __syncthreads();
  }

  int b = bh / H, h = bh % H;
  float liA = 1.f / lA;
  float liB = 1.f / lB;
  bf16_t* obA = outT + (((size_t)b * Tq + (t0 + nn))      * (H * CH)) + h * CH;
  bf16_t* obB = outT + (((size_t)b * Tq + (t0 + 16 + nn)) * (H * CH)) + h * CH;
#pragma unroll
  for (int r = 0; r < 8; r++) {
    int m = r + 8 * laneHi;
    obA[m]      = f2bf(oaA0[r] * liA);
    obA[16 + m] = f2bf(oaA1[r] * liA);
    obA[32 + m] = f2bf(oaA2[r] * liA);
    obA[48 + m] = f2bf(oaA3[r] * liA);
    obB[m]      = f2bf(oaB0[r] * liB);
    obB[16 + m] = f2bf(oaB1[r] * liB);
    obB[32 + m] = f2bf(oaB2[r] * liB);
    obB[48 + m] = f2bf(oaB3[r] * liB);
  }
}

// ---------------------------------------------------------------------------
// Host launcher
// ---------------------------------------------------------------------------
extern "C" void kernel_launch(void* const* d_in, const int* in_sizes, int n_in,
                              void* d_out, int out_size, void* d_ws, size_t ws_size,
                              hipStream_t stream) {
  (void)in_sizes; (void)n_in; (void)out_size; (void)ws_size;
  const int B = 8, C = 512, T = 1024, T2 = 1024, H = 8, CH = 64, G = 32;
  const int Mqkv = 3 * C, Mkv = 2 * C;
  const float scale = 0.3535533905932738f;   // 1/sqrt(sqrt(64))

  const float* x       = (const float*)d_in[0];
  const float* cond2   = (const float*)d_in[1];
  const float* norm_g  = (const float*)d_in[2];
  const float* norm_b  = (const float*)d_in[3];
  const float* qkv_w   = (const float*)d_in[4];
  const float* qkv_b   = (const float*)d_in[5];
  const float* proj_w  = (const float*)d_in[6];
  const float* proj_b  = (const float*)d_in[7];
  const float* c2_w    = (const float*)d_in[8];
  const float* c2_b    = (const float*)d_in[9];
  const float* ncg     = (const float*)d_in[10];
  const float* ncb     = (const float*)d_in[11];
  const float* ckv_w   = (const float*)d_in[12];
  const float* ckv_b   = (const float*)d_in[13];
  const float* cproj_w = (const float*)d_in[14];
  const float* cproj_b = (const float*)d_in[15];
  float* out = (float*)d_out;

  char* ws = (char*)d_ws;
  size_t off = 0;
  auto walloc = [&](size_t bytes) -> void* {
    void* p = ws + off;
    off += (bytes + 255) & ~(size_t)255;
    return p;
  };
  bf16_t* wqkv  = (bf16_t*)walloc((size_t)Mqkv * C * 2);
  bf16_t* wproj = (bf16_t*)walloc((size_t)C * C * 2);
  bf16_t* wkv   = (bf16_t*)walloc((size_t)Mkv * C * 2);
  bf16_t* wcpj  = (bf16_t*)walloc((size_t)C * C * 2);
  bf16_t* xnT   = (bf16_t*)walloc((size_t)B * T * C * 2);      // reused for cond branch
  bf16_t* qTb   = (bf16_t*)walloc((size_t)B * H * T * CH * 2); // reused for cross Q
  bf16_t* kTb   = (bf16_t*)walloc((size_t)B * H * T * CH * 2); // reused for cross K
  bf16_t* vBuf  = (bf16_t*)walloc((size_t)B * H * CH * T * 2); // reused for cross V
  bf16_t* oT    = (bf16_t*)walloc((size_t)B * T * C * 2);      // attn out (both)
  float*  hself = (float*)walloc((size_t)B * C * T * 4);
  float*  c2p   = (float*)walloc((size_t)B * C * T2 * 4);

  // Weight fp32 -> bf16
  f32_to_bf16_kernel<<<(Mqkv * C + 255) / 256, 256, 0, stream>>>(qkv_w,  wqkv,  Mqkv * C);
  f32_to_bf16_kernel<<<(C * C + 255) / 256,   256, 0, stream>>>(proj_w, wproj, C * C);
  f32_to_bf16_kernel<<<(Mkv * C + 255) / 256, 256, 0, stream>>>(ckv_w,  wkv,   Mkv * C);
  f32_to_bf16_kernel<<<(C * C + 255) / 256,   256, 0, stream>>>(cproj_w, wcpj, C * C);

  // ---- self attention path ----
  gn_transpose_kernel<<<B * G, 256, 0, stream>>>(x, norm_g, norm_b, xnT, C, T, G);
  gemm_split_kernel<<<dim3(T / 64, Mqkv / 16, B), 32, 0, stream>>>(
      wqkv, xnT, qkv_b, qTb, kTb, vBuf, Mqkv, C, T, CH, scale, 0);
  flash_attn_kernel<<<dim3(T / 32, B * H), 32, 0, stream>>>(
      qTb, kTb, vBuf, oT, T, T, H);
  gemm_resid_kernel<<<dim3(T / 64, C / 16, B), 32, 0, stream>>>(
      wproj, oT, proj_b, x, hself, qTb, C, C, T, CH, scale);  // also emits cross-Q

  // ---- cross attention path ----
  cond2_proj_kernel<<<(int)(((size_t)B * C * T2 + 255) / 256), 256, 0, stream>>>(
      cond2, c2_w, c2_b, c2p, C, T2, B);
  gn_transpose_kernel<<<B * G, 256, 0, stream>>>(c2p, ncg, ncb, xnT, C, T2, G);
  gemm_split_kernel<<<dim3(T2 / 64, Mkv / 16, B), 32, 0, stream>>>(
      wkv, xnT, ckv_b, nullptr, kTb, vBuf, Mkv, C, T2, CH, scale, 1);
  flash_attn_kernel<<<dim3(T / 32, B * H), 32, 0, stream>>>(
      qTb, kTb, vBuf, oT, T, T2, H);
  gemm_resid_kernel<<<dim3(T / 64, C / 16, B), 32, 0, stream>>>(
      wcpj, oT, cproj_b, hself, out, nullptr, C, C, T, CH, scale);
}
